// HyperLSTMCell_14980845928863
// MI455X (gfx1250) — matrix operand order repaired and checked by reading, not compiled, not run on hardware
//
#include <hip/hip_runtime.h>
#include <hip/hip_bf16.h>
#include <math.h>

#define TB 256

typedef __bf16 bf16_t;
typedef __attribute__((ext_vector_type(16))) __bf16 v16bf;
typedef __attribute__((ext_vector_type(8)))  float  v8f;

static constexpr int Bn = 16384, Dn = 128, Hn = 512, HHn = 256, En = 32;
static constexpr float kForget = 1.0f, kEps = 1e-5f;

static __device__ __forceinline__ v8f vzero() { v8f z = {}; return z; }

static __device__ __forceinline__ v8f wmma_bf16(v16bf a, v16bf b, v8f c) {
  // D = A(16x32 bf16) * B(32x16 bf16) + C(16x16 f32)
  return __builtin_amdgcn_wmma_f32_16x16x32_bf16(false, a, false, b, (short)0, c, false, false);
}

// A fragment (16x32 bf16), row-major source (LDS or global), leading dim lda.
// ISA 7.12.2: lane m=lane&15, hi=lane>>4 holds K = 8*hi+{0..7} and 16+8*hi+{0..7}.
static __device__ __forceinline__ v16bf load_a(const bf16_t* A, int lda, int k0, int lane) {
  const int m = lane & 15, hi = lane >> 4;
  const bf16_t* p = A + m * lda + k0 + 8 * hi;
  v16bf a;
#pragma unroll
  for (int i = 0; i < 8; ++i) { a[i] = p[i]; a[8 + i] = p[16 + i]; }
  return a;
}

// B fragment (32x16 bf16) from weights stored (N x K) row-major (= column-major B = W^T).
static __device__ __forceinline__ v16bf load_b(const bf16_t* __restrict__ W, int ldw,
                                               int n0, int k0, int lane) {
  const int n = n0 + (lane & 15), hi = lane >> 4;
  const bf16_t* q = W + (size_t)n * ldw + k0 + 8 * hi;
  v16bf b;
#pragma unroll
  for (int i = 0; i < 8; ++i) { b[i] = q[i]; b[8 + i] = q[16 + i]; }
  return b;
}

// Butterfly sum over the 16 lanes of this lane's half-wave.
static __device__ __forceinline__ float half_sum(float v) {
  v += __shfl_xor(v, 1, 32);
  v += __shfl_xor(v, 2, 32);
  v += __shfl_xor(v, 4, 32);
  v += __shfl_xor(v, 8, 32);
  return v;
}

static __device__ __forceinline__ float sigmoidf(float x) { return 1.0f / (1.0f + __expf(-x)); }

// gfx1250 async copy: global -> LDS, 16B per lane, tracked by ASYNCcnt.
static __device__ __forceinline__ void async_b128(void* lds_dst, const void* gsrc) {
  asm volatile("global_load_async_to_lds_b128 %0, %1, off"
               :
               : "v"((unsigned)(uintptr_t)lds_dst), "v"(gsrc)
               : "memory");
}
static __device__ __forceinline__ void wait_async0() {
  asm volatile("s_wait_asynccnt 0" ::: "memory");
}

template <int ROWS, int COLS, int LDP>
static __device__ __forceinline__ void stage_async(bf16_t* lds, const bf16_t* __restrict__ src,
                                                   long rowbase) {
  constexpr int CH = COLS / 8;  // 16B chunks per row
  for (int idx = threadIdx.x; idx < ROWS * CH; idx += TB) {
    const int r = idx / CH, c = idx % CH;
    async_b128(&lds[r * LDP + c * 8], &src[(rowbase + r) * (long)COLS + c * 8]);
  }
}

// ---------------- casts ----------------

__global__ __launch_bounds__(TB) void k_cast(const float* __restrict__ src,
                                             bf16_t* __restrict__ dst,
                                             int rows, int cols, int ld) {
  const long n = (long)rows * cols;
  for (long i = (long)blockIdx.x * TB + threadIdx.x; i < n; i += (long)gridDim.x * TB) {
    const long r = i / cols, c = i - r * cols;
    dst[i] = (bf16_t)src[r * ld + c];
  }
}

__global__ __launch_bounds__(TB) void k_copy(const float* __restrict__ src,
                                             float* __restrict__ dst, int n) {
  const int i = blockIdx.x * TB + threadIdx.x;
  if (i < n) dst[i] = src[i];
}

// ---------------- hyper LSTM step: 32 rows/block, B-frag register reuse ----------------

__global__ __launch_bounds__(TB, 1) void k_hyper(
    const bf16_t* __restrict__ xb, const bf16_t* __restrict__ hhob,
    const bf16_t* __restrict__ hwih, const bf16_t* __restrict__ hwhh,
    const float* __restrict__ lnh_w, const float* __restrict__ lnh_b,
    const float* __restrict__ lnc_w, const float* __restrict__ lnc_b,
    const float* __restrict__ c_total,
    float* __restrict__ out_ht, float* __restrict__ out_ct, bf16_t* __restrict__ hhnb) {
  __shared__ __align__(16) bf16_t xs[2][16 * 136];
  __shared__ __align__(16) bf16_t hs[2][16 * 264];
  __shared__ float redS[4 * 32 * 8], redQ[4 * 32 * 8];
  __shared__ float mu_s[128], rs_s[128];
  __shared__ float crS[32 * 8], crQ[32 * 8];
  __shared__ float cmu[32], crsd[32];

  const int tid = threadIdx.x, lane = tid & 31, wv = tid >> 5;
  const int nl = lane & 15, hi = lane >> 4;
  const long rowbase = (long)blockIdx.x * 32;

  stage_async<16, 128, 136>(xs[0], xb, rowbase);
  stage_async<16, 128, 136>(xs[1], xb, rowbase + 16);
  stage_async<16, 256, 264>(hs[0], hhob, rowbase);
  stage_async<16, 256, 264>(hs[1], hhob, rowbase + 16);
  wait_async0();
  __syncthreads();

  // wave wv owns units [wv*32, wv*32+32) for all 4 gates, two M-tiles.
  v8f acc[4][2][2];
#pragma unroll
  for (int g = 0; g < 4; ++g) {
    const int cg = g * 256 + wv * 32;
#pragma unroll
    for (int mt = 0; mt < 2; ++mt)
#pragma unroll
      for (int t = 0; t < 2; ++t) acc[g][mt][t] = vzero();
#pragma unroll
    for (int k0 = 0; k0 < 128; k0 += 32) {
      v16bf bfr[2];
#pragma unroll
      for (int t = 0; t < 2; ++t) bfr[t] = load_b(hwih, 128, cg + t * 16, k0, lane);
#pragma unroll
      for (int mt = 0; mt < 2; ++mt) {
        const v16bf a = load_a(xs[mt], 136, k0, lane);
#pragma unroll
        for (int t = 0; t < 2; ++t) acc[g][mt][t] = wmma_bf16(a, bfr[t], acc[g][mt][t]);
      }
    }
#pragma unroll
    for (int k0 = 0; k0 < 256; k0 += 32) {
      v16bf bfr[2];
#pragma unroll
      for (int t = 0; t < 2; ++t) bfr[t] = load_b(hwhh, 256, cg + t * 16, k0, lane);
#pragma unroll
      for (int mt = 0; mt < 2; ++mt) {
        const v16bf a = load_a(hs[mt], 264, k0, lane);
#pragma unroll
        for (int t = 0; t < 2; ++t) acc[g][mt][t] = wmma_bf16(a, bfr[t], acc[g][mt][t]);
      }
    }
  }

  // chunk-LN stats per (gate, row) over 256 units
#pragma unroll
  for (int g = 0; g < 4; ++g)
#pragma unroll
    for (int mt = 0; mt < 2; ++mt) {
      float ps[8], pq[8];
#pragma unroll
      for (int r = 0; r < 8; ++r) { ps[r] = 0.f; pq[r] = 0.f; }
#pragma unroll
      for (int t = 0; t < 2; ++t)
#pragma unroll
        for (int r = 0; r < 8; ++r) { const float v = acc[g][mt][t][r]; ps[r] += v; pq[r] += v * v; }
#pragma unroll
      for (int r = 0; r < 8; ++r) { ps[r] = half_sum(ps[r]); pq[r] = half_sum(pq[r]); }
      if (nl == 0) {
#pragma unroll
        for (int r = 0; r < 8; ++r) {
          const int rw = mt * 16 + r + 8 * hi;
          redS[(g * 32 + rw) * 8 + wv] = ps[r];
          redQ[(g * 32 + rw) * 8 + wv] = pq[r];
        }
      }
    }
  __syncthreads();
  if (tid < 128) {
    float s = 0.f, q = 0.f;
#pragma unroll
    for (int w = 0; w < 8; ++w) { s += redS[tid * 8 + w]; q += redQ[tid * 8 + w]; }
    const float m = s * (1.0f / 256.0f);
    mu_s[tid] = m;
    rs_s[tid] = rsqrtf(q * (1.0f / 256.0f) - m * m + kEps);
  }
  __syncthreads();

#pragma unroll
  for (int g = 0; g < 4; ++g)
#pragma unroll
    for (int mt = 0; mt < 2; ++mt)
#pragma unroll
      for (int t = 0; t < 2; ++t) {
        const int col = g * 256 + wv * 32 + t * 16 + nl;
#pragma unroll
        for (int r = 0; r < 8; ++r) {
          const int rw = mt * 16 + r + 8 * hi;
          const float xn = (acc[g][mt][t][r] - mu_s[g * 32 + rw]) * rs_s[g * 32 + rw];
          const float lv = xn * lnh_w[col] + lnh_b[col];
          float a;
          if (g == 1)      a = sigmoidf(lv + kForget);
          else if (g == 3) a = tanhf(lv);
          else             a = sigmoidf(lv);
          acc[g][mt][t][r] = a;
        }
      }

  // c update + row-LN over 256
  v8f cc[2][2];
#pragma unroll
  for (int mt = 0; mt < 2; ++mt) {
    float cps[8], cpq[8];
#pragma unroll
    for (int r = 0; r < 8; ++r) { cps[r] = 0.f; cpq[r] = 0.f; }
#pragma unroll
    for (int t = 0; t < 2; ++t)
#pragma unroll
      for (int r = 0; r < 8; ++r) {
        const int rw = mt * 16 + r + 8 * hi;
        const int u = wv * 32 + t * 16 + nl;
        const float cold = c_total[(rowbase + rw) * 768 + 512 + u];
        const float cn = acc[1][mt][t][r] * cold + acc[0][mt][t][r] * acc[3][mt][t][r];
        cc[mt][t][r] = cn;
        cps[r] += cn; cpq[r] += cn * cn;
      }
#pragma unroll
    for (int r = 0; r < 8; ++r) { cps[r] = half_sum(cps[r]); cpq[r] = half_sum(cpq[r]); }
    if (nl == 0)
#pragma unroll
      for (int r = 0; r < 8; ++r) {
        const int rw = mt * 16 + r + 8 * hi;
        crS[rw * 8 + wv] = cps[r]; crQ[rw * 8 + wv] = cpq[r];
      }
  }
  __syncthreads();
  if (tid < 32) {
    float s = 0.f, q = 0.f;
#pragma unroll
    for (int w = 0; w < 8; ++w) { s += crS[tid * 8 + w]; q += crQ[tid * 8 + w]; }
    const float m = s * (1.0f / 256.0f);
    cmu[tid] = m; crsd[tid] = rsqrtf(q * (1.0f / 256.0f) - m * m + kEps);
  }
  __syncthreads();
#pragma unroll
  for (int mt = 0; mt < 2; ++mt)
#pragma unroll
    for (int t = 0; t < 2; ++t)
#pragma unroll
      for (int r = 0; r < 8; ++r) {
        const int rw = mt * 16 + r + 8 * hi;
        const int u = wv * 32 + t * 16 + nl;
        const long grow = rowbase + rw;
        const float cn = cc[mt][t][r];
        const float hv = acc[2][mt][t][r] * tanhf((cn - cmu[rw]) * crsd[rw] * lnc_w[u] + lnc_b[u]);
        out_ht[grow * 768 + 512 + u] = hv;
        out_ct[grow * 768 + 512 + u] = cn;
        hhnb[grow * 256 + u] = (bf16_t)hv;
      }
}

// ---------------- z = h_hyper @ ZwCat^T + bias (bf16 out), 32 rows/block ----------------

__global__ __launch_bounds__(TB, 1) void k_z(
    const bf16_t* __restrict__ hhnb, const bf16_t* __restrict__ zwcat,
    const float* __restrict__ zbias, bf16_t* __restrict__ zout) {
  __shared__ __align__(16) bf16_t hs[2][16 * 264];
  const int tid = threadIdx.x, lane = tid & 31, wv = tid >> 5;
  const int nl = lane & 15, hi = lane >> 4;
  const long rowbase = (long)blockIdx.x * 32;
  stage_async<16, 256, 264>(hs[0], hhnb, rowbase);
  stage_async<16, 256, 264>(hs[1], hhnb, rowbase + 16);
  wait_async0();
  __syncthreads();
  v8f acc[2][3];
#pragma unroll
  for (int mt = 0; mt < 2; ++mt)
#pragma unroll
    for (int t = 0; t < 3; ++t) acc[mt][t] = vzero();
#pragma unroll
  for (int k0 = 0; k0 < 256; k0 += 32) {
    v16bf bfr[3];
#pragma unroll
    for (int t = 0; t < 3; ++t) bfr[t] = load_b(zwcat, 256, wv * 48 + t * 16, k0, lane);
#pragma unroll
    for (int mt = 0; mt < 2; ++mt) {
      const v16bf a = load_a(hs[mt], 264, k0, lane);
#pragma unroll
      for (int t = 0; t < 3; ++t) acc[mt][t] = wmma_bf16(a, bfr[t], acc[mt][t]);
    }
  }
#pragma unroll
  for (int mt = 0; mt < 2; ++mt)
#pragma unroll
    for (int t = 0; t < 3; ++t)
#pragma unroll
      for (int r = 0; r < 8; ++r) {
        const int rw = mt * 16 + r + 8 * hi;
        const int j = wv * 48 + t * 16 + nl;
        zout[(rowbase + rw) * 384 + j] = (bf16_t)(acc[mt][t][r] + zbias[j]);
      }
}

// ---------------- main fused gates: 32 rows/block ----------------

__global__ __launch_bounds__(TB, 1) void k_main(
    const bf16_t* __restrict__ xb, const bf16_t* __restrict__ hb, const bf16_t* __restrict__ zb,
    const bf16_t* __restrict__ wih, const bf16_t* __restrict__ whh,
    const bf16_t* __restrict__ axb, const bf16_t* __restrict__ ahb, const bf16_t* __restrict__ bhb,
    const float* __restrict__ lnh_w, const float* __restrict__ lnh_b,
    const float* __restrict__ lnc_w, const float* __restrict__ lnc_b,
    const float* __restrict__ c_total,
    float* __restrict__ out_h, float* __restrict__ out_ht, float* __restrict__ out_ct) {
  __shared__ __align__(16) bf16_t xs[2][16 * 136];
  __shared__ __align__(16) bf16_t hs[2][16 * 520];
  __shared__ float redS[4 * 32 * 8], redQ[4 * 32 * 8];
  __shared__ float mu_s[128], rs_s[128];
  __shared__ float crS[32 * 8], crQ[32 * 8];
  __shared__ float cmu[32], crsd[32];

  const int tid = threadIdx.x, lane = tid & 31, wv = tid >> 5;
  const int nl = lane & 15, hi = lane >> 4;
  const long rowbase = (long)blockIdx.x * 32;

  stage_async<16, 128, 136>(xs[0], xb, rowbase);
  stage_async<16, 128, 136>(xs[1], xb, rowbase + 16);
  stage_async<16, 512, 520>(hs[0], hb, rowbase);
  stage_async<16, 512, 520>(hs[1], hb, rowbase + 16);
  wait_async0();
  __syncthreads();

  // wave wv owns units [wv*64, wv*64+64) for all 4 gates, two M-tiles.
  v8f acc[4][2][4];
#pragma unroll
  for (int g = 0; g < 4; ++g) {
    const int cgW = g * 512 + wv * 64;  // row into W_ih / W_hh
    const int cgU = wv * 64;            // row into alpha[g]/beta[g]

    // Wx accumulated directly into acc[g]
#pragma unroll
    for (int mt = 0; mt < 2; ++mt)
#pragma unroll
      for (int t = 0; t < 4; ++t) acc[g][mt][t] = vzero();
#pragma unroll
    for (int k0 = 0; k0 < 128; k0 += 32) {
      v16bf bfr[4];
#pragma unroll
      for (int t = 0; t < 4; ++t) bfr[t] = load_b(wih, 128, cgW + t * 16, k0, lane);
#pragma unroll
      for (int mt = 0; mt < 2; ++mt) {
        const v16bf a = load_a(xs[mt], 136, k0, lane);
#pragma unroll
        for (int t = 0; t < 4; ++t) acc[g][mt][t] = wmma_bf16(a, bfr[t], acc[g][mt][t]);
      }
    }
    // Sx (K=32), consumed tile-at-a-time: acc = Wx .* Sx
    {
      v16bf bfa[4];
#pragma unroll
      for (int t = 0; t < 4; ++t) bfa[t] = load_b(axb + (size_t)g * 512 * 32, 32, cgU + t * 16, 0, lane);
#pragma unroll
      for (int mt = 0; mt < 2; ++mt) {
        const v16bf az = load_a(zb + (rowbase + mt * 16) * 384, 384, g * 32, lane);
#pragma unroll
        for (int t = 0; t < 4; ++t) {
          const v8f s = wmma_bf16(az, bfa[t], vzero());
          acc[g][mt][t] = acc[g][mt][t] * s;
        }
      }
    }
    // Wh into tw
    v8f tw[2][4];
#pragma unroll
    for (int mt = 0; mt < 2; ++mt)
#pragma unroll
      for (int t = 0; t < 4; ++t) tw[mt][t] = vzero();
#pragma unroll
    for (int k0 = 0; k0 < 512; k0 += 32) {
      __builtin_prefetch((const void*)(whh + (size_t)(cgW + nl) * 512 + k0 + 128), 0, 1);
      v16bf bfr[4];
#pragma unroll
      for (int t = 0; t < 4; ++t) bfr[t] = load_b(whh, 512, cgW + t * 16, k0, lane);
#pragma unroll
      for (int mt = 0; mt < 2; ++mt) {
        const v16bf a = load_a(hs[mt], 520, k0, lane);
#pragma unroll
        for (int t = 0; t < 4; ++t) tw[mt][t] = wmma_bf16(a, bfr[t], tw[mt][t]);
      }
    }
    // Sh: acc += Wh .* Sh
    {
      v16bf bfa[4];
#pragma unroll
      for (int t = 0; t < 4; ++t) bfa[t] = load_b(ahb + (size_t)g * 512 * 32, 32, cgU + t * 16, 0, lane);
#pragma unroll
      for (int mt = 0; mt < 2; ++mt) {
        const v16bf az = load_a(zb + (rowbase + mt * 16) * 384, 384, 128 + g * 32, lane);
#pragma unroll
        for (int t = 0; t < 4; ++t) {
          const v8f s = wmma_bf16(az, bfa[t], vzero());
          acc[g][mt][t] += tw[mt][t] * s;
        }
      }
    }
    // Bh: acc += Bh
    {
      v16bf bfa[4];
#pragma unroll
      for (int t = 0; t < 4; ++t) bfa[t] = load_b(bhb + (size_t)g * 512 * 32, 32, cgU + t * 16, 0, lane);
#pragma unroll
      for (int mt = 0; mt < 2; ++mt) {
        const v16bf az = load_a(zb + (rowbase + mt * 16) * 384, 384, 256 + g * 32, lane);
#pragma unroll
        for (int t = 0; t < 4; ++t) {
          const v8f s = wmma_bf16(az, bfa[t], vzero());
          acc[g][mt][t] += s;
        }
      }
    }
  }

  // chunk-LN stats per (gate,row) over 512 units
#pragma unroll
  for (int g = 0; g < 4; ++g)
#pragma unroll
    for (int mt = 0; mt < 2; ++mt) {
      float ps[8], pq[8];
#pragma unroll
      for (int r = 0; r < 8; ++r) { ps[r] = 0.f; pq[r] = 0.f; }
#pragma unroll
      for (int t = 0; t < 4; ++t)
#pragma unroll
        for (int r = 0; r < 8; ++r) { const float v = acc[g][mt][t][r]; ps[r] += v; pq[r] += v * v; }
#pragma unroll
      for (int r = 0; r < 8; ++r) { ps[r] = half_sum(ps[r]); pq[r] = half_sum(pq[r]); }
      if (nl == 0) {
#pragma unroll
        for (int r = 0; r < 8; ++r) {
          const int rw = mt * 16 + r + 8 * hi;
          redS[(g * 32 + rw) * 8 + wv] = ps[r];
          redQ[(g * 32 + rw) * 8 + wv] = pq[r];
        }
      }
    }
  __syncthreads();
  if (tid < 128) {
    float s = 0.f, q = 0.f;
#pragma unroll
    for (int w = 0; w < 8; ++w) { s += redS[tid * 8 + w]; q += redQ[tid * 8 + w]; }
    const float m = s * (1.0f / 512.0f);
    mu_s[tid] = m;
    rs_s[tid] = rsqrtf(q * (1.0f / 512.0f) - m * m + kEps);
  }
  __syncthreads();

#pragma unroll
  for (int g = 0; g < 4; ++g)
#pragma unroll
    for (int mt = 0; mt < 2; ++mt)
#pragma unroll
      for (int t = 0; t < 4; ++t) {
        const int col = g * 512 + wv * 64 + t * 16 + nl;
#pragma unroll
        for (int r = 0; r < 8; ++r) {
          const int rw = mt * 16 + r + 8 * hi;
          const float xn = (acc[g][mt][t][r] - mu_s[g * 32 + rw]) * rs_s[g * 32 + rw];
          const float lv = xn * lnh_w[col] + lnh_b[col];
          float a;
          if (g == 1)      a = sigmoidf(lv + kForget);
          else if (g == 3) a = tanhf(lv);
          else             a = sigmoidf(lv);
          acc[g][mt][t][r] = a;
        }
      }

  // c update + row-LN over 512
  v8f cc[2][4];
#pragma unroll
  for (int mt = 0; mt < 2; ++mt) {
    float cps[8], cpq[8];
#pragma unroll
    for (int r = 0; r < 8; ++r) { cps[r] = 0.f; cpq[r] = 0.f; }
#pragma unroll
    for (int t = 0; t < 4; ++t)
#pragma unroll
      for (int r = 0; r < 8; ++r) {
        const int rw = mt * 16 + r + 8 * hi;
        const int u = wv * 64 + t * 16 + nl;
        const float cold = c_total[(rowbase + rw) * 768 + u];
        const float cn = acc[1][mt][t][r] * cold + acc[0][mt][t][r] * acc[3][mt][t][r];
        cc[mt][t][r] = cn;
        cps[r] += cn; cpq[r] += cn * cn;
      }
#pragma unroll
    for (int r = 0; r < 8; ++r) { cps[r] = half_sum(cps[r]); cpq[r] = half_sum(cpq[r]); }
    if (nl == 0)
#pragma unroll
      for (int r = 0; r < 8; ++r) {
        const int rw = mt * 16 + r + 8 * hi;
        crS[rw * 8 + wv] = cps[r]; crQ[rw * 8 + wv] = cpq[r];
      }
  }
  __syncthreads();
  if (tid < 32) {
    float s = 0.f, q = 0.f;
#pragma unroll
    for (int w = 0; w < 8; ++w) { s += crS[tid * 8 + w]; q += crQ[tid * 8 + w]; }
    const float m = s * (1.0f / 512.0f);
    cmu[tid] = m; crsd[tid] = rsqrtf(q * (1.0f / 512.0f) - m * m + kEps);
  }
  __syncthreads();
#pragma unroll
  for (int mt = 0; mt < 2; ++mt)
#pragma unroll
    for (int t = 0; t < 4; ++t)
#pragma unroll
      for (int r = 0; r < 8; ++r) {
        const int rw = mt * 16 + r + 8 * hi;
        const int u = wv * 64 + t * 16 + nl;
        const long grow = rowbase + rw;
        const float cn = cc[mt][t][r];
        const float hv = acc[2][mt][t][r] * tanhf((cn - cmu[rw]) * crsd[rw] * lnc_w[u] + lnc_b[u]);
        out_h[grow * 512 + u] = hv;
        out_ht[grow * 768 + u] = hv;
        out_ct[grow * 768 + u] = cn;
      }
}

// ---------------- launch ----------------

extern "C" void kernel_launch(void* const* d_in, const int* in_sizes, int n_in,
                              void* d_out, int out_size, void* d_ws, size_t ws_size,
                              hipStream_t stream) {
  const float* x         = (const float*)d_in[0];
  const float* h_total   = (const float*)d_in[1];
  const float* c_total   = (const float*)d_in[2];
  const float* W_ih      = (const float*)d_in[3];
  const float* W_hh      = (const float*)d_in[4];
  const float* lnh_w     = (const float*)d_in[5];
  const float* lnh_b     = (const float*)d_in[6];
  const float* lnc_w     = (const float*)d_in[7];
  const float* lnc_b     = (const float*)d_in[8];
  const float* hyp_Wih   = (const float*)d_in[9];
  const float* hyp_Whh   = (const float*)d_in[10];
  const float* hyp_lnh_w = (const float*)d_in[11];
  const float* hyp_lnh_b = (const float*)d_in[12];
  const float* hyp_lnc_w = (const float*)d_in[13];
  const float* hyp_lnc_b = (const float*)d_in[14];
  const float* zw_x_w    = (const float*)d_in[15];
  const float* zw_x_b    = (const float*)d_in[16];
  const float* alpha_x   = (const float*)d_in[17];
  const float* zw_h_w    = (const float*)d_in[18];
  const float* zw_h_b    = (const float*)d_in[19];
  const float* alpha_h   = (const float*)d_in[20];
  const float* zb_h_w    = (const float*)d_in[21];
  const float* zb_h_b    = (const float*)d_in[22];
  const float* beta_h    = (const float*)d_in[23];
  (void)in_sizes; (void)n_in; (void)out_size; (void)ws_size;

  float* out_h  = (float*)d_out;
  float* out_ht = out_h + (size_t)Bn * Hn;
  float* out_ct = out_ht + (size_t)Bn * (Hn + HHn);

  char* wsb = (char*)d_ws;
  size_t off = 0;
  auto take = [&](size_t bytes) {
    char* p = wsb + off;
    off = (off + bytes + 255) & ~(size_t)255;
    return p;
  };
  bf16_t* xb    = (bf16_t*)take((size_t)Bn * Dn * 2);
  bf16_t* hb    = (bf16_t*)take((size_t)Bn * Hn * 2);
  bf16_t* hhob  = (bf16_t*)take((size_t)Bn * HHn * 2);
  bf16_t* hhnb  = (bf16_t*)take((size_t)Bn * HHn * 2);
  bf16_t* zbuf  = (bf16_t*)take((size_t)Bn * 384 * 2);
  bf16_t* wihb  = (bf16_t*)take((size_t)4 * Hn * Dn * 2);
  bf16_t* whhb  = (bf16_t*)take((size_t)4 * Hn * Hn * 2);
  bf16_t* hwihb = (bf16_t*)take((size_t)4 * HHn * Dn * 2);
  bf16_t* hwhhb = (bf16_t*)take((size_t)4 * HHn * HHn * 2);
  bf16_t* axb   = (bf16_t*)take((size_t)4 * Hn * En * 2);
  bf16_t* ahb   = (bf16_t*)take((size_t)4 * Hn * En * 2);
  bf16_t* bhb   = (bf16_t*)take((size_t)4 * Hn * En * 2);
  bf16_t* zwc   = (bf16_t*)take((size_t)384 * HHn * 2);
  float*  zbias = (float*)take((size_t)384 * 4);

  auto CG = [](long n) { long b = (n + TB - 1) / TB; return (int)(b > 4096 ? 4096 : b); };

  k_cast<<<CG((long)Bn * Dn), TB, 0, stream>>>(x, xb, Bn, Dn, Dn);
  k_cast<<<CG((long)Bn * Hn), TB, 0, stream>>>(h_total, hb, Bn, Hn, Hn + HHn);
  k_cast<<<CG((long)Bn * HHn), TB, 0, stream>>>(h_total + Hn, hhob, Bn, HHn, Hn + HHn);
  k_cast<<<CG(4L * Hn * Dn), TB, 0, stream>>>(W_ih, wihb, 4 * Hn, Dn, Dn);
  k_cast<<<CG(4L * Hn * Hn), TB, 0, stream>>>(W_hh, whhb, 4 * Hn, Hn, Hn);
  k_cast<<<CG(4L * HHn * Dn), TB, 0, stream>>>(hyp_Wih, hwihb, 4 * HHn, Dn, Dn);
  k_cast<<<CG(4L * HHn * HHn), TB, 0, stream>>>(hyp_Whh, hwhhb, 4 * HHn, HHn, HHn);
  k_cast<<<CG(4L * Hn * En), TB, 0, stream>>>(alpha_x, axb, 4 * Hn, En, En);
  k_cast<<<CG(4L * Hn * En), TB, 0, stream>>>(alpha_h, ahb, 4 * Hn, En, En);
  k_cast<<<CG(4L * Hn * En), TB, 0, stream>>>(beta_h, bhb, 4 * Hn, En, En);
  k_cast<<<CG(128L * HHn), TB, 0, stream>>>(zw_x_w, zwc, 128, HHn, HHn);
  k_cast<<<CG(128L * HHn), TB, 0, stream>>>(zw_h_w, zwc + 128 * HHn, 128, HHn, HHn);
  k_cast<<<CG(128L * HHn), TB, 0, stream>>>(zb_h_w, zwc + 256 * HHn, 128, HHn, HHn);
  k_copy<<<1, TB, 0, stream>>>(zw_x_b, zbias, 128);
  k_copy<<<1, TB, 0, stream>>>(zw_h_b, zbias + 128, 128);
  k_copy<<<1, TB, 0, stream>>>(zb_h_b, zbias + 256, 128);

  k_hyper<<<Bn / 32, TB, 0, stream>>>(xb, hhob, hwihb, hwhhb, hyp_lnh_w, hyp_lnh_b,
                                      hyp_lnc_w, hyp_lnc_b, c_total, out_ht, out_ct, hhnb);
  k_z<<<Bn / 32, TB, 0, stream>>>(hhnb, zwc, zbias, zbuf);
  k_main<<<Bn / 32, TB, 0, stream>>>(xb, hb, zbuf, wihb, whhb, axb, ahb, bhb,
                                     lnh_w, lnh_b, lnc_w, lnc_b, c_total,
                                     out_h, out_ht, out_ct);
}
